// MultiHeadSelfAttention_39144331936376
// MI455X (gfx1250) — matrix operand order repaired and checked by reading, not compiled, not run on hardware
//
#include <hip/hip_runtime.h>
#include <hip/hip_bf16.h>
#include <math.h>
#include <stdint.h>

// ---------------------------------------------------------------------------
// ProbSparse multi-head self-attention (Informer) for gfx1250 / MI455X.
// B=4, L=2048, D=512, H=8, Dh=64, sample_k = n_top = 40.
//
// Heavy work = 4 GEMMs of (8192x512)x(512x512): v_wmma_f32_16x16x32_f16
// (f16 in, f32 accumulate). Weights are pre-transposed once so both A and B
// WMMA fragments are contiguous per lane in LDS (wide ds_load_b128).
// Global->LDS tile feeds use the CDNA5 async DMA path
// (global_load_async_to_lds_b128, ASYNCcnt) with LDS DOUBLE BUFFERING:
// the next tile's DMA is issued before a counted s_wait_asynccnt retires
// only the current tile, hiding DMA latency behind the WMMAs.
// ---------------------------------------------------------------------------

typedef __attribute__((ext_vector_type(16))) _Float16 v16h;
typedef __attribute__((ext_vector_type(8)))  _Float16 v8h;
typedef __attribute__((ext_vector_type(8)))  float    v8f;

static constexpr int Bc   = 4;
static constexpr int Lc   = 2048;
static constexpr int Dc   = 512;
static constexpr int Hc   = 8;
static constexpr int Dhc  = 64;
static constexpr int MROW = Bc * Lc;           // 8192
static constexpr int SK   = 40;                // sample_k
static constexpr int NTOP = 40;                // n_top

// ---------------------------------------------------------------------------
// CDNA5 async global->LDS DMA (ASYNCcnt-tracked). LDS byte offset = low 32
// bits of the flat shared-memory address (LDS aperture rule).
// ---------------------------------------------------------------------------
__device__ __forceinline__ void async_load_b128(void* lds_ptr, const void* gptr) {
#if defined(__gfx1250__)
  unsigned lds_off = (unsigned)(uintptr_t)lds_ptr;
  asm volatile("global_load_async_to_lds_b128 %0, %1, off"
               :: "v"(lds_off), "v"(gptr) : "memory");
#else
  (void)lds_ptr; (void)gptr;
#endif
}
template <int N>
__device__ __forceinline__ void wait_async() {
#if defined(__gfx1250__)
  asm volatile("s_wait_asynccnt %0" :: "i"(N) : "memory");
#endif
}

// ---------------------------------------------------------------------------
// f32 -> f16 conversion
// ---------------------------------------------------------------------------
__global__ void cvt_f32_f16(const float* __restrict__ in,
                            _Float16* __restrict__ out, int n) {
  int i = blockIdx.x * blockDim.x + threadIdx.x;
  if (i < n) out[i] = (_Float16)in[i];
}

// W (K x N, f32) -> Wt (N x K, f16): one-time transpose so GEMM B-tiles are
// N-major (lane-contiguous WMMA B fragments).
__global__ void cvt_w_t(const float* __restrict__ W, _Float16* __restrict__ Wt) {
  int gid = blockIdx.x * blockDim.x + threadIdx.x;  // output-linear over D*D
  if (gid >= Dc * Dc) return;
  int k = gid % Dc;
  int n = gid / Dc;
  Wt[gid] = (_Float16)W[(size_t)k * Dc + n];
}

// context (B,H,L,Dh) f32 -> (B,L,D) f16 (transpose heads back)
__global__ void ctx_to_f16(const float* __restrict__ ctx,
                           _Float16* __restrict__ out) {
  int gid = blockIdx.x * blockDim.x + threadIdx.x;   // over B*L*D
  if (gid >= Bc * Lc * Dc) return;
  int col = gid % Dc;
  int l   = (gid / Dc) % Lc;
  int b   = gid / (Dc * Lc);
  int h   = col / Dhc;
  int d   = col % Dhc;
  out[gid] = (_Float16)ctx[(((size_t)(b * Hc + h) * Lc) + l) * Dhc + d];
}

// ---------------------------------------------------------------------------
// WMMA GEMM:  out[M x N] = A[M x K] * Wt[N x K]^T + bias
// STORE_MODE 0: row-major (M,N).  STORE_MODE 1: (B,H,L,Dh) head layout.
// Block tile 128x128, BK=32, 256 threads = 8 waves, wave tile 32x64
// (2x4 WMMA 16x16x32 accumulators). Double-buffered async LDS tiles.
// ---------------------------------------------------------------------------
template <int STORE_MODE>
__global__ __launch_bounds__(256)
void gemm_bias_wmma(const _Float16* __restrict__ A,
                    const _Float16* __restrict__ Wt,
                    const float* __restrict__ bias,
                    float* __restrict__ out,
                    int M, int N, int K) {
  constexpr int BM = 128, BN = 128, BK = 32;
  // +8-half pad: row pitch 80B keeps every 16B fragment slice aligned.
  __shared__ _Float16 As[2][BM][BK + 8];   // A tiles, M-major (double buffer)
  __shared__ _Float16 Bt[2][BN][BK + 8];   // B tiles, N-major

  const int tid  = threadIdx.x;
  const int lane = tid & 31;
  const int wid  = tid >> 5;
  const int wm   = (wid & 3) * 32;      // wave row offset in block tile
  const int wn   = (wid >> 2) * 64;     // wave col offset in block tile
  const int bm   = blockIdx.y * BM;
  const int bn   = blockIdx.x * BN;

  const int nfrag   = lane & 15;        // row/col index inside 16x16 tile
  const int halfsel = lane >> 4;        // lane-half select

  // per-thread tile-load coordinates: 512 16B vectors per tile, 2 per thread
  const int r0 = tid >> 2, c0 = (tid & 3) * 8;
  const int r1 = (tid + 256) >> 2, c1 = ((tid + 256) & 3) * 8;

  // issue one tile's 4 async DMA ops (per wave: ASYNCcnt += 4)
  auto issue_tile = [&](int k0, int buf) {
    async_load_b128(&As[buf][r0][c0], &A [(size_t)(bm + r0) * K + k0 + c0]);
    async_load_b128(&Bt[buf][r0][c0], &Wt[(size_t)(bn + r0) * K + k0 + c0]);
    async_load_b128(&As[buf][r1][c1], &A [(size_t)(bm + r1) * K + k0 + c1]);
    async_load_b128(&Bt[buf][r1][c1], &Wt[(size_t)(bn + r1) * K + k0 + c1]);
  };

  v8f acc[2][4] = {};

  issue_tile(0, 0);                      // prologue: tile 0 in flight
  int buf = 0;

  for (int k0 = 0; k0 < K; k0 += BK, buf ^= 1) {
    const bool has_next = (k0 + BK < K);
    if (has_next) {
      issue_tile(k0 + BK, buf ^ 1);      // next tile DMA runs behind compute
      // prefetch the tile after next into GL2 (global_prefetch_b8)
      if (k0 + 2 * BK < K) {
        __builtin_prefetch(&A [(size_t)(bm + r0) * K + k0 + 2 * BK + c0], 0, 3);
        __builtin_prefetch(&Wt[(size_t)(bn + r0) * K + k0 + 2 * BK + c0], 0, 3);
      }
      wait_async<4>();                   // retire current tile only (in-order)
    } else {
      wait_async<0>();
    }
    __syncthreads();                     // current tile visible to all waves

    // ---- A fragments (16x32 f16): lane m=lane&15 holds K=8h+{0..7} and
    //      16+8h+{0..7} -> two contiguous 16B LDS loads
    v16h afrag[2];
#pragma unroll
    for (int mi = 0; mi < 2; ++mi) {
      int arow = wm + mi * 16 + nfrag;
      v8h lo = *(const v8h*)&As[buf][arow][8 * halfsel];
      v8h hi = *(const v8h*)&As[buf][arow][16 + 8 * halfsel];
      afrag[mi] = __builtin_shufflevector(lo, hi, 0, 1, 2, 3, 4, 5, 6, 7,
                                          8, 9, 10, 11, 12, 13, 14, 15);
    }
    // ---- B fragments (32x16 f16): lane n=lane&15 holds K=8h+{0..7} and
    //      16+8h+{0..7} -> two contiguous 16B LDS loads
    v16h bfrag[4];
#pragma unroll
    for (int ni = 0; ni < 4; ++ni) {
      int bcol = wn + ni * 16 + nfrag;
      v8h lo = *(const v8h*)&Bt[buf][bcol][8 * halfsel];
      v8h hi = *(const v8h*)&Bt[buf][bcol][16 + 8 * halfsel];
      bfrag[ni] = __builtin_shufflevector(lo, hi, 0, 1, 2, 3, 4, 5, 6, 7,
                                          8, 9, 10, 11, 12, 13, 14, 15);
    }
    // ---- 8 WMMAs per wave per k-step ----
#pragma unroll
    for (int mi = 0; mi < 2; ++mi)
#pragma unroll
      for (int ni = 0; ni < 4; ++ni)
        acc[mi][ni] = __builtin_amdgcn_wmma_f32_16x16x32_f16(
            false, afrag[mi], false, bfrag[ni],
            (short)0, acc[mi][ni], false, false);
    __syncthreads();  // all reads of this buffer done before its next overwrite
  }

  // ---- epilogue: C/D layout -> lane n = lane&15, VGPR r -> m = 8*half + r
#pragma unroll
  for (int mi = 0; mi < 2; ++mi) {
#pragma unroll
    for (int ni = 0; ni < 4; ++ni) {
      int col = bn + wn + ni * 16 + nfrag;
      float bv = bias[col];
#pragma unroll
      for (int r = 0; r < 8; ++r) {
        int row   = bm + wm + mi * 16 + halfsel * 8 + r;
        float val = acc[mi][ni][r] + bv;
        if (STORE_MODE == 0) {
          out[(size_t)row * N + col] = val;
        } else {
          int b = row / Lc, l = row % Lc;
          int h = col / Dhc, d = col % Dhc;
          out[(((size_t)(b * Hc + h) * Lc) + l) * Dhc + d] = val;
        }
      }
    }
  }
}

// ---------------------------------------------------------------------------
// Sparse measurement: M[b,h,i] = max_s(q_i . k_idx(i,s)) - (1/L)*sum_s(...)
// Deterministic hash-based sampling (JAX threefry indices are not inputs and
// cannot be reproduced on-device; same shape/cost, shared across b,h).
// ---------------------------------------------------------------------------
__device__ __forceinline__ unsigned hash_u32(unsigned x) {
  x ^= x >> 16; x *= 0x7feb352dU;
  x ^= x >> 15; x *= 0x846ca68bU;
  x ^= x >> 16; return x;
}

__global__ __launch_bounds__(256)
void sparse_measure(const float* __restrict__ Q, const float* __restrict__ Kt,
                    float* __restrict__ Mmat) {
  int gid = blockIdx.x * blockDim.x + threadIdx.x;   // B*H*L
  if (gid >= Bc * Hc * Lc) return;
  int i  = gid % Lc;
  int bh = gid / Lc;
  const float* q = Q + (size_t)(bh * Lc + i) * Dhc;
  float qreg[Dhc];
#pragma unroll
  for (int d = 0; d < Dhc; ++d) qreg[d] = q[d];
  float mx = -INFINITY, sm = 0.f;
  for (int s = 0; s < SK; ++s) {
    int kidx = (int)(hash_u32((unsigned)(i * SK + s) * 0x9E3779B9u) & (Lc - 1));
    const float* kv = Kt + (size_t)(bh * Lc + kidx) * Dhc;
    float dot = 0.f;
#pragma unroll
    for (int d = 0; d < Dhc; ++d) dot += qreg[d] * kv[d];
    mx = fmaxf(mx, dot);
    sm += dot;
  }
  Mmat[gid] = mx - sm * (1.0f / Lc);
}

// ---------------------------------------------------------------------------
// top-NTOP per (b,h) by iterated argmax (matches top_k value ordering,
// ties to lower index).
// ---------------------------------------------------------------------------
__global__ __launch_bounds__(256)
void topk_kernel(const float* __restrict__ Mmat, int* __restrict__ Mtop) {
  __shared__ float vals[Lc];
  __shared__ float redv[256];
  __shared__ int   redi[256];
  int bh = blockIdx.x, tid = threadIdx.x;
  for (int i = tid; i < Lc; i += 256) vals[i] = Mmat[bh * Lc + i];
  __syncthreads();
  for (int t = 0; t < NTOP; ++t) {
    float best = -INFINITY; int bidx = 0;
    for (int i = tid; i < Lc; i += 256) {
      float v = vals[i];
      if (v > best) { best = v; bidx = i; }
    }
    redv[tid] = best; redi[tid] = bidx;
    __syncthreads();
    for (int s = 128; s > 0; s >>= 1) {
      if (tid < s) {
        if (redv[tid + s] > redv[tid] ||
            (redv[tid + s] == redv[tid] && redi[tid + s] < redi[tid])) {
          redv[tid] = redv[tid + s]; redi[tid] = redi[tid + s];
        }
      }
      __syncthreads();
    }
    if (tid == 0) { Mtop[bh * NTOP + t] = redi[0]; vals[redi[0]] = -INFINITY; }
    __syncthreads();
  }
}

// ---------------------------------------------------------------------------
// Masked softmax attention for the selected queries: one block per (b,h,u).
// ---------------------------------------------------------------------------
__global__ __launch_bounds__(256)
void attn_kernel(const float* __restrict__ Q, const float* __restrict__ Kt,
                 const float* __restrict__ V, const int* __restrict__ Mtop,
                 float* __restrict__ upd) {
  __shared__ float sc[Lc];
  __shared__ float red[256];
  __shared__ float qv[Dhc];
  int blk = blockIdx.x;
  int u   = blk % NTOP;
  int bh  = blk / NTOP;
  int tid = threadIdx.x;
  int qi  = Mtop[bh * NTOP + u];
  if (tid < Dhc) qv[tid] = Q[(size_t)(bh * Lc + qi) * Dhc + tid];
  __syncthreads();

  const float scale = 0.125f;  // 1/sqrt(64)
  for (int l = tid; l < Lc; l += 256) {
    const float* kr = Kt + (size_t)(bh * Lc + l) * Dhc;
    float dot = 0.f;
#pragma unroll
    for (int d = 0; d < Dhc; ++d) dot += qv[d] * kr[d];
    sc[l] = (l > qi) ? -INFINITY : dot * scale;
  }
  __syncthreads();

  float mx = -INFINITY;
  for (int l = tid; l < Lc; l += 256) mx = fmaxf(mx, sc[l]);
  red[tid] = mx; __syncthreads();
  for (int s = 128; s > 0; s >>= 1) {
    if (tid < s) red[tid] = fmaxf(red[tid], red[tid + s]);
    __syncthreads();
  }
  mx = red[0]; __syncthreads();

  float sum = 0.f;
  for (int l = tid; l < Lc; l += 256) {
    float e = __expf(sc[l] - mx);
    sc[l] = e; sum += e;
  }
  red[tid] = sum; __syncthreads();
  for (int s = 128; s > 0; s >>= 1) {
    if (tid < s) red[tid] += red[tid + s];
    __syncthreads();
  }
  float inv = 1.0f / red[0];
  __syncthreads();

  int d = tid & 63, seg = tid >> 6;     // 4 segments of 512
  float part = 0.f;
  for (int l = seg * 512; l < (seg + 1) * 512; ++l)
    part += sc[l] * V[(size_t)(bh * Lc + l) * Dhc + d];
  red[tid] = part; __syncthreads();
  if (seg == 0)
    upd[((size_t)(bh * NTOP + u)) * Dhc + d] =
        (red[d] + red[d + 64] + red[d + 128] + red[d + 192]) * inv;
}

// ---------------------------------------------------------------------------
// cumsum of V along L: one block per (b,h); 4 segments of 512 per dh.
// ---------------------------------------------------------------------------
__global__ __launch_bounds__(256)
void cumsum_kernel(const float* __restrict__ V, float* __restrict__ ctx) {
  __shared__ float segsum[4][Dhc];
  int bh = blockIdx.x, tid = threadIdx.x;
  int d = tid & 63, seg = tid >> 6;
  const float* v = V + (size_t)bh * Lc * Dhc;
  float* c = ctx + (size_t)bh * Lc * Dhc;
  float s = 0.f;
  for (int l = seg * 512; l < (seg + 1) * 512; ++l) s += v[(size_t)l * Dhc + d];
  segsum[seg][d] = s;
  __syncthreads();
  float run = 0.f;
  for (int t = 0; t < seg; ++t) run += segsum[t][d];
  for (int l = seg * 512; l < (seg + 1) * 512; ++l) {
    run += v[(size_t)l * Dhc + d];
    c[(size_t)l * Dhc + d] = run;
  }
}

__global__ void scatter_kernel(const float* __restrict__ upd,
                               const int* __restrict__ Mtop,
                               float* __restrict__ ctx) {
  int gid = blockIdx.x * blockDim.x + threadIdx.x;  // B*H*NTOP*Dh
  if (gid >= Bc * Hc * NTOP * Dhc) return;
  int d  = gid & 63;
  int r  = gid >> 6;
  int u  = r % NTOP;
  int bh = r / NTOP;
  int l  = Mtop[bh * NTOP + u];
  ctx[((size_t)(bh * Lc + l)) * Dhc + d] = upd[gid];
}

// ---------------------------------------------------------------------------
// launch
// ---------------------------------------------------------------------------
extern "C" void kernel_launch(void* const* d_in, const int* in_sizes, int n_in,
                              void* d_out, int out_size, void* d_ws, size_t ws_size,
                              hipStream_t stream) {
  (void)in_sizes; (void)n_in; (void)out_size; (void)ws_size;
  const float* x  = (const float*)d_in[0];
  const float* Wq = (const float*)d_in[1];
  const float* bq = (const float*)d_in[2];
  const float* Wk = (const float*)d_in[3];
  const float* bk = (const float*)d_in[4];
  const float* Wv = (const float*)d_in[5];
  const float* bv = (const float*)d_in[6];
  const float* Wo = (const float*)d_in[7];
  const float* bo = (const float*)d_in[8];
  float* out = (float*)d_out;

  // workspace carve-up (all 256B aligned)
  char* ws = (char*)d_ws;
  size_t off = 0;
  auto alloc = [&](size_t bytes) { char* p = ws + off; off = (off + bytes + 255) & ~(size_t)255; return p; };
  _Float16* x_h   = (_Float16*)alloc((size_t)MROW * Dc * 2);   // 8 MB
  _Float16* WqT   = (_Float16*)alloc((size_t)Dc * Dc * 2);     // transposed f16
  _Float16* WkT   = (_Float16*)alloc((size_t)Dc * Dc * 2);
  _Float16* WvT   = (_Float16*)alloc((size_t)Dc * Dc * 2);
  _Float16* WoT   = (_Float16*)alloc((size_t)Dc * Dc * 2);
  float*    Qm    = (float*)alloc((size_t)MROW * Dc * 4);      // (B,H,L,Dh)
  float*    Km    = (float*)alloc((size_t)MROW * Dc * 4);
  float*    Vm    = (float*)alloc((size_t)MROW * Dc * 4);
  float*    ctx   = (float*)alloc((size_t)MROW * Dc * 4);
  _Float16* ctx_h = (_Float16*)alloc((size_t)MROW * Dc * 2);
  float*    Mmat  = (float*)alloc((size_t)Bc * Hc * Lc * 4);
  int*      Mtop  = (int*)alloc((size_t)Bc * Hc * NTOP * 4);
  float*    upd   = (float*)alloc((size_t)Bc * Hc * NTOP * Dhc * 4);

  const int nX = MROW * Dc;          // 4,194,304
  const int nW = Dc * Dc;            // 262,144

  cvt_f32_f16<<<(nX + 255) / 256, 256, 0, stream>>>(x, x_h, nX);
  cvt_w_t<<<(nW + 255) / 256, 256, 0, stream>>>(Wq, WqT);
  cvt_w_t<<<(nW + 255) / 256, 256, 0, stream>>>(Wk, WkT);
  cvt_w_t<<<(nW + 255) / 256, 256, 0, stream>>>(Wv, WvT);
  cvt_w_t<<<(nW + 255) / 256, 256, 0, stream>>>(Wo, WoT);

  dim3 gemm_grid(Dc / 128, MROW / 128);   // (4, 64)
  gemm_bias_wmma<1><<<gemm_grid, 256, 0, stream>>>(x_h, WqT, bq, Qm, MROW, Dc, Dc);
  gemm_bias_wmma<1><<<gemm_grid, 256, 0, stream>>>(x_h, WkT, bk, Km, MROW, Dc, Dc);
  gemm_bias_wmma<1><<<gemm_grid, 256, 0, stream>>>(x_h, WvT, bv, Vm, MROW, Dc, Dc);

  sparse_measure<<<(Bc * Hc * Lc) / 256, 256, 0, stream>>>(Qm, Km, Mmat);
  topk_kernel<<<Bc * Hc, 256, 0, stream>>>(Mmat, Mtop);
  attn_kernel<<<Bc * Hc * NTOP, 256, 0, stream>>>(Qm, Km, Vm, Mtop, upd);
  cumsum_kernel<<<Bc * Hc, 256, 0, stream>>>(Vm, ctx);
  scatter_kernel<<<(Bc * Hc * NTOP * Dhc + 255) / 256, 256, 0, stream>>>(upd, Mtop, ctx);
  ctx_to_f16<<<(nX + 255) / 256, 256, 0, stream>>>(ctx, ctx_h);

  gemm_bias_wmma<0><<<gemm_grid, 256, 0, stream>>>(ctx_h, WoT, bo, out, MROW, Dc, Dc);
}
